// MemAttention_64939905516066
// MI455X (gfx1250) — compile-verified, matrix operand order checked by reading, not run on hardware
//
#include <hip/hip_runtime.h>
#include <hip/hip_fp16.h>

#define L_SEQ 2048
#define N_BATCH 4
#define E_DIM 1024
#define H_HEADS 16
#define D_HEAD 64
#define LAM 0.001f
#define LN_EPS 1e-5f

typedef _Float16 half16 __attribute__((ext_vector_type(16)));
typedef _Float16 half8v __attribute__((ext_vector_type(8)));
typedef float float8 __attribute__((ext_vector_type(8)));
typedef float float4v __attribute__((ext_vector_type(4)));
typedef unsigned uint4v __attribute__((ext_vector_type(4)));
typedef int int4v __attribute__((ext_vector_type(4)));
typedef int int8v __attribute__((ext_vector_type(8)));

#if __has_builtin(__builtin_amdgcn_tensor_load_to_lds)
#define HAVE_TDM 1
#endif

// K-index map for 16-bit A/B fragments (ISA 7.12.2, 16-bit A 16x32 table).
__device__ __forceinline__ int kmap(int h, int g) {
  int j = h >> 1, lo = h & 1;
  int hi = j >> 2, jj = j & 3;
  return hi * 16 + g * 8 + jj * 2 + lo;
}

__device__ __forceinline__ float8 wmma16(half16 a, half16 b, float8 c) {
  return __builtin_amdgcn_wmma_f32_16x16x32_f16(false, a, false, b,
                                                (short)0, c, false, false);
}

__device__ __forceinline__ float gelu_exact(float x) {
  return 0.5f * x * (1.0f + erff(x * 0.70710678118654752440f));
}

// CDNA5 async global->LDS copy, 16B per lane, tracked by ASYNCcnt.
__device__ __forceinline__ void async_copy_b128(unsigned lds_off,
                                                const void* gptr) {
  asm volatile("global_load_async_to_lds_b128 %0, %1, off"
               :
               : "v"(lds_off), "v"((unsigned long long)(size_t)gptr)
               : "memory");
}
__device__ __forceinline__ void wait_async0() {
  asm volatile("s_wait_asynccnt 0x0" ::: "memory");
}
__device__ __forceinline__ unsigned lds_addr(const void* p) {
  return (unsigned)(size_t)p;  // low 32 bits of shared aperture = LDS offset
}

#ifdef HAVE_TDM
// TDM: DMA one 2D tile (32 rows x 128B, row stride 2048B) global -> LDS.
// D# group0: count=1, lds_addr, 57b global addr, type=2.
// D# group1: data_size=8B, tile 16x32 (8B units), tensor_dim0_stride=256.
__device__ __forceinline__ void tdm_load_tile_32x128(unsigned lds_off,
                                                     const void* gptr) {
  unsigned long long ga = (unsigned long long)(size_t)gptr;
  uint4v g0;
  g0[0] = 1u;                    // count=1, is_restore=0, gather=0
  g0[1] = lds_off;               // LDS byte address
  g0[2] = (unsigned)ga;          // global_addr[31:0]
  g0[3] = ((unsigned)(ga >> 32) & 0x01FFFFFFu) | 0x80000000u;  // addr|type=2
  int8v g1;
  g1[0] = 3 << 16;               // workgroup_mask=0, data_size=3 (8B)
  g1[1] = 16 << 16;              // tensor_dim0 = 16 (8B units = 128B)
  g1[2] = 32 << 16;              // tensor_dim0 hi=0 | tensor_dim1 = 32
  g1[3] = 16 << 16;              // tensor_dim1 hi=0 | tile_dim0 = 16
  g1[4] = 32;                    // tile_dim1 = 32, tile_dim2 = 0
  g1[5] = 256;                   // tensor_dim0_stride = 256 (8B units = 2048B)
  g1[6] = 0;
  g1[7] = 0;
  int4v z4 = {0, 0, 0, 0};
#if __has_include(<hip/amd_detail/amd_gfx1250_TDM.h>)
  int8v z8 = {0, 0, 0, 0, 0, 0, 0, 0};
  __builtin_amdgcn_tensor_load_to_lds(g0, g1, z4, z4, z8, 0);
#else
  __builtin_amdgcn_tensor_load_to_lds(g0, g1, z4, z4, 0);
#endif
}
#endif

// ---------------------------------------------------------------------------
// Kernel 0: vectorized f32 -> f16 conversion (8 elements / thread).
// ---------------------------------------------------------------------------
__global__ __launch_bounds__(256) void cvt_f16_kernel(
    const float* __restrict__ in, _Float16* __restrict__ out, int n8) {
  int i = blockIdx.x * 256 + threadIdx.x;
  if (i >= n8) return;
  const float4v* p = (const float4v*)in + (size_t)i * 2;
  float4v a = p[0], b = p[1];
  half8v h;
  h[0] = (_Float16)a[0]; h[1] = (_Float16)a[1];
  h[2] = (_Float16)a[2]; h[3] = (_Float16)a[3];
  h[4] = (_Float16)b[0]; h[5] = (_Float16)b[1];
  h[6] = (_Float16)b[2]; h[7] = (_Float16)b[3];
  *((half8v*)out + i) = h;
}

// ---------------------------------------------------------------------------
// Kernel 1: act = gelu(X @ W^T + bias), X pre-converted f16 (M=8192 x E).
// 128x128 tile / 8-wave WG; K-slabs double-buffered via async LDS copies.
// ---------------------------------------------------------------------------
__global__ __launch_bounds__(256) void proj_gelu_kernel(
    const _Float16* __restrict__ X16, const _Float16* __restrict__ W16,
    const float* __restrict__ bias, const float* __restrict__ memory,
    _Float16* __restrict__ act_out, _Float16* __restrict__ mem_out,
    int do_mem) {
  __shared__ __align__(16) _Float16 Xs[2][128 * 32];
  __shared__ __align__(16) _Float16 Ws[2][128 * 32];
  const int tid = threadIdx.x;
  const int lane = tid & 31, wid = tid >> 5;
  const int g = lane >> 4, ln = lane & 15;
  const int m0 = blockIdx.y * 128;
  const int e0 = blockIdx.x * 128;

  auto stage = [&](int kc, int buf) {
    #pragma unroll
    for (int i = 0; i < 2; ++i) {
      int idx = tid + i * 256;  // 0..511 : row = idx>>2, seg = idx&3
      int r = idx >> 2, seg = idx & 3;
      async_copy_b128(lds_addr(&Xs[buf][0]) + idx * 16,
                      (const char*)(X16 + (size_t)(m0 + r) * E_DIM + kc) + seg * 16);
      async_copy_b128(lds_addr(&Ws[buf][0]) + idx * 16,
                      (const char*)(W16 + (size_t)(e0 + r) * E_DIM + kc) + seg * 16);
    }
  };

  float8 acc[8] = {};
  stage(0, 0);
  wait_async0();
  __syncthreads();

  int buf = 0;
  for (int kc = 0; kc < E_DIM; kc += 32, buf ^= 1) {
    if (kc + 32 < E_DIM) stage(kc + 32, buf ^ 1);  // prefetch next slab

    half16 a;
    #pragma unroll
    for (int h = 0; h < 16; ++h)
      a[h] = Xs[buf][(wid * 16 + ln) * 32 + kmap(h, g)];
    #pragma unroll
    for (int t = 0; t < 8; ++t) {
      half16 b;
      #pragma unroll
      for (int h = 0; h < 16; ++h)
        b[h] = Ws[buf][(t * 16 + ln) * 32 + kmap(h, g)];
      acc[t] = wmma16(a, b, acc[t]);
    }
    wait_async0();
    __syncthreads();
  }

  #pragma unroll
  for (int t = 0; t < 8; ++t) {
    int e = e0 + t * 16 + ln;
    float bv = bias[e];
    #pragma unroll
    for (int r = 0; r < 8; ++r) {
      int m = m0 + wid * 16 + r + 8 * g;  // C layout: M = r + 8*(lane>=16)
      int l = m >> 2, n = m & 3;          // rows are (l*N + n)
      float v = gelu_exact(acc[t][r] + bv);
      size_t oi = ((size_t)(n * L_SEQ + l)) * E_DIM + e;
      act_out[oi] = (_Float16)v;
      if (do_mem) {
        float mv = LAM * memory[(size_t)l * E_DIM + e] + (1.0f - LAM) * v;
        mem_out[oi] = (_Float16)mv;
      }
    }
  }
}

// ---------------------------------------------------------------------------
// Kernel 2: per (batch*head) causal linear attention (zero-masked, no
// softmax): w = qh @ kh^T (s<=l), out += w @ mh. K/mem chunks staged by the
// Tensor Data Mover (TENSORcnt), issued by wave 0 only (TDM ignores EXEC),
// double-buffered against WMMA compute.
// ---------------------------------------------------------------------------
__global__ __launch_bounds__(128) void attn_kernel(
    const _Float16* __restrict__ qf,   // (N,L,E)
    const _Float16* __restrict__ kf,   // (N,L,E)
    const _Float16* __restrict__ mf,   // (N,L,E)
    float* __restrict__ out) {         // (L,N,E) pre-LN
  __shared__ __align__(16) _Float16 khs[2][32 * 64];
  __shared__ __align__(16) _Float16 mhs[2][32 * 64];
  __shared__ float scs[4][16 * 32];    // per-wave score staging (m x s_local)
  const int tid = threadIdx.x;
  const int lane = tid & 31, wid = tid >> 5;
  const int g = lane >> 4, ln = lane & 15;
  const int l0 = blockIdx.x * 64;
  const int b = blockIdx.y;            // n*H + h
  const int n = b >> 4, hh = b & 15;
  const size_t base = ((size_t)n * L_SEQ) * E_DIM + hh * D_HEAD;
  // scalar wave id -> true scalar branch for single-wave TDM issue
  const int swid = __builtin_amdgcn_readfirstlane(wid);

  auto stage = [&](int sb, int buf) {
#ifdef HAVE_TDM
    if (swid == 0) {
      tdm_load_tile_32x128(lds_addr(&khs[buf][0]),
                           kf + base + (size_t)sb * E_DIM);
      tdm_load_tile_32x128(lds_addr(&mhs[buf][0]),
                           mf + base + (size_t)sb * E_DIM);
    }
#else
    #pragma unroll
    for (int i = 0; i < 2; ++i) {
      int idx = tid + i * 128;  // 0..255 : row = idx>>3, seg = idx&7
      int r = idx >> 3, seg = idx & 7;
      async_copy_b128(lds_addr(&khs[buf][0]) + idx * 16,
                      (const char*)(kf + base + (size_t)(sb + r) * E_DIM) + seg * 16);
      async_copy_b128(lds_addr(&mhs[buf][0]) + idx * 16,
                      (const char*)(mf + base + (size_t)(sb + r) * E_DIM) + seg * 16);
    }
#endif
  };
  auto wait_stage = [&]() {
#ifdef HAVE_TDM
    if (swid == 0) __builtin_amdgcn_s_wait_tensorcnt(0);
#else
    wait_async0();
#endif
    __syncthreads();
  };

  // preload q A-fragments for this wave's 16 rows (K = d = 64 -> 2 chunks)
  half16 aq[2];
  {
    const _Float16* qrow = qf + base + (size_t)(l0 + wid * 16 + ln) * E_DIM;
    #pragma unroll
    for (int cc = 0; cc < 2; ++cc)
      #pragma unroll
      for (int h = 0; h < 16; ++h) aq[cc][h] = qrow[cc * 32 + kmap(h, g)];
  }

  float8 oacc[4] = {};
  const int s_end = l0 + 64;           // causal truncation
  stage(0, 0);
  wait_stage();

  int buf = 0;
  for (int sb = 0; sb < s_end; sb += 32, buf ^= 1) {
    if (sb + 32 < s_end) stage(sb + 32, buf ^ 1);  // prefetch next chunk

    // scores: 16 rows x 32 s, K = 64 (two chunks)
    float8 cs0 = {}, cs1 = {};
    #pragma unroll
    for (int cc = 0; cc < 2; ++cc) {
      half16 b0, b1;
      #pragma unroll
      for (int h = 0; h < 16; ++h) {
        int k = kmap(h, g);
        b0[h] = khs[buf][(0 * 16 + ln) * 64 + cc * 32 + k];
        b1[h] = khs[buf][(1 * 16 + ln) * 64 + cc * 32 + k];
      }
      cs0 = wmma16(aq[cc], b0, cs0);
      cs1 = wmma16(aq[cc], b1, cs1);
    }

    // causal mask (zero, not -inf) + stage to per-wave LDS row-major
    #pragma unroll
    for (int r = 0; r < 8; ++r) {
      int m = r + 8 * g;
      int lg = l0 + wid * 16 + m;
      scs[wid][m * 32 + ln]      = (sb + ln      <= lg) ? cs0[r] : 0.0f;
      scs[wid][m * 32 + 16 + ln] = (sb + 16 + ln <= lg) ? cs1[r] : 0.0f;
    }
    asm volatile("s_wait_dscnt 0x0" ::: "memory");  // wave-local LDS RAW

    // re-read scores in A-fragment layout (f32 -> f16)
    half16 a2;
    #pragma unroll
    for (int h = 0; h < 16; ++h)
      a2[h] = (_Float16)scs[wid][ln * 32 + kmap(h, g)];

    // out(16x64) += scores(16x32) @ mh(32x64)
    #pragma unroll
    for (int t = 0; t < 4; ++t) {
      half16 bm;
      #pragma unroll
      for (int h = 0; h < 16; ++h)
        bm[h] = mhs[buf][kmap(h, g) * 64 + t * 16 + ln];
      oacc[t] = wmma16(a2, bm, oacc[t]);
    }
    wait_stage();
  }

  // store pre-LN output (L,N,E) f32
  #pragma unroll
  for (int t = 0; t < 4; ++t)
    #pragma unroll
    for (int r = 0; r < 8; ++r) {
      int l = l0 + wid * 16 + r + 8 * g;
      int e = hh * D_HEAD + t * 16 + ln;
      out[((size_t)(l * N_BATCH + n)) * E_DIM + e] = oacc[t][r];
    }
}

// ---------------------------------------------------------------------------
// Kernel 3: LayerNorm over E=1024 per (l,n) row.
// ---------------------------------------------------------------------------
__global__ __launch_bounds__(256) void ln_kernel(
    const float* __restrict__ x, const float* __restrict__ w,
    const float* __restrict__ bb, float* __restrict__ out) {
  __shared__ float s1[256], s2[256];
  const int row = blockIdx.x;
  const float* xr = x + (size_t)row * E_DIM;
  float v[4];
  float sum = 0.f, sq = 0.f;
  #pragma unroll
  for (int i = 0; i < 4; ++i) {
    v[i] = xr[threadIdx.x + i * 256];
    sum += v[i];
    sq += v[i] * v[i];
  }
  s1[threadIdx.x] = sum;
  s2[threadIdx.x] = sq;
  __syncthreads();
  for (int off = 128; off > 0; off >>= 1) {
    if (threadIdx.x < off) {
      s1[threadIdx.x] += s1[threadIdx.x + off];
      s2[threadIdx.x] += s2[threadIdx.x + off];
    }
    __syncthreads();
  }
  float mean = s1[0] * (1.0f / E_DIM);
  float var = s2[0] * (1.0f / E_DIM) - mean * mean;
  float rstd = rsqrtf(var + LN_EPS);
  #pragma unroll
  for (int i = 0; i < 4; ++i) {
    int e = threadIdx.x + i * 256;
    out[(size_t)row * E_DIM + e] = (v[i] - mean) * rstd * w[e] + bb[e];
  }
}

extern "C" void kernel_launch(void* const* d_in, const int* in_sizes, int n_in,
                              void* d_out, int out_size, void* d_ws,
                              size_t ws_size, hipStream_t stream) {
  (void)in_sizes; (void)n_in; (void)out_size; (void)ws_size;
  const float* query  = (const float*)d_in[0];
  const float* key    = (const float*)d_in[1];
  /* d_in[2] = value: unused by the reference */
  const float* Wq     = (const float*)d_in[3];
  const float* bq     = (const float*)d_in[4];
  const float* Wk     = (const float*)d_in[5];
  const float* bk     = (const float*)d_in[6];
  const float* memory = (const float*)d_in[7];
  const float* ln_w   = (const float*)d_in[8];
  const float* ln_b   = (const float*)d_in[9];
  float* out = (float*)d_out;

  char* ws = (char*)d_ws;
  const size_t szAct = (size_t)N_BATCH * L_SEQ * E_DIM * sizeof(_Float16); // 16 MB
  const size_t szW   = (size_t)E_DIM * E_DIM * sizeof(_Float16);           // 2 MB
  _Float16* x16q = (_Float16*)ws; ws += szAct;   // query as f16 (M x E)
  _Float16* x16k = (_Float16*)ws; ws += szAct;   // key as f16
  _Float16* w16q = (_Float16*)ws; ws += szW;
  _Float16* w16k = (_Float16*)ws; ws += szW;
  _Float16* qf   = (_Float16*)ws; ws += szAct;   // gelu(qWq^T+b) (N,L,E)
  _Float16* kf   = (_Float16*)ws; ws += szAct;
  _Float16* mf   = (_Float16*)ws; ws += szAct;
  float* attn    = (float*)ws;                   // 32 MB (L,N,E)

  const int nAct8 = (L_SEQ * N_BATCH * E_DIM) / 8;  // 1M
  const int nW8   = (E_DIM * E_DIM) / 8;            // 128K
  hipLaunchKernelGGL(cvt_f16_kernel, dim3(nAct8 / 256), dim3(256), 0, stream,
                     query, x16q, nAct8);
  hipLaunchKernelGGL(cvt_f16_kernel, dim3(nAct8 / 256), dim3(256), 0, stream,
                     key, x16k, nAct8);
  hipLaunchKernelGGL(cvt_f16_kernel, dim3(nW8 / 256), dim3(256), 0, stream,
                     Wq, w16q, nW8);
  hipLaunchKernelGGL(cvt_f16_kernel, dim3(nW8 / 256), dim3(256), 0, stream,
                     Wk, w16k, nW8);

  dim3 gProj(E_DIM / 128, (L_SEQ * N_BATCH) / 128);
  hipLaunchKernelGGL(proj_gelu_kernel, gProj, dim3(256), 0, stream,
                     x16q, w16q, bq, memory, qf, mf, 1);
  hipLaunchKernelGGL(proj_gelu_kernel, gProj, dim3(256), 0, stream,
                     x16k, w16k, bk, memory, kf, mf, 0);
  hipLaunchKernelGGL(attn_kernel, dim3(L_SEQ / 64, N_BATCH * H_HEADS),
                     dim3(128), 0, stream, qf, kf, mf, attn);
  hipLaunchKernelGGL(ln_kernel, dim3(L_SEQ * N_BATCH), dim3(256), 0, stream,
                     attn, ln_w, ln_b, out);
}
